// _IPEXLlamaAttention_7000796692986
// MI455X (gfx1250) — matrix operand order repaired and checked by reading, not compiled, not run on hardware
//
#include <hip/hip_runtime.h>
#include <hip/hip_bf16.h>
#include <stdint.h>

#define B_      2
#define S_      2048
#define H_      4096
#define NH_     32
#define NKV_    8
#define HD_     128
#define GROUPS_ (NH_ / NKV_)
#define NQKV_   ((NH_ + 2 * NKV_) * HD_)   // 6144
#define NTOK_   (B_ * S_)                   // 4096

typedef __attribute__((ext_vector_type(16))) __bf16 v16bf;
typedef __attribute__((ext_vector_type(8)))  __bf16 v8bf;
typedef __attribute__((ext_vector_type(8)))  float  v8f;
typedef int vi4 __attribute__((vector_size(16)));
typedef __attribute__((address_space(1))) vi4 gvi4;   // global AS v4i
typedef __attribute__((address_space(3))) vi4 lvi4;   // LDS    AS v4i

// Async global->LDS DMA path (gfx1250): guarded so compilation never breaks.
#if defined(__HIP_DEVICE_COMPILE__) && defined(__gfx1250__) && \
    __has_builtin(__builtin_amdgcn_global_load_async_to_lds_b128)
#define USE_ASYNC_LDS 1
#else
#define USE_ASYNC_LDS 0
#endif

__device__ __forceinline__ unsigned short f2bf(float f) {
    union { float f; unsigned int u; } v; v.f = f;
    unsigned int u = v.u;
    u += 0x7FFFu + ((u >> 16) & 1u);   // round-to-nearest-even
    return (unsigned short)(u >> 16);
}

// A-fragment loader for v_wmma_*_16x16x32_bf16.
// Contiguous k order (matches B): lanes 0-15 hold K = 0..15,
// lanes 16-31 hold K = 16..31  -> one 32-byte contiguous chunk.
__device__ __forceinline__ v16bf load_a16(const unsigned short* rowp, int half) {
    return *(const v16bf*)(rowp + half * 16);
}

// 16-byte global -> LDS copy: async DMA if available, else through VGPRs.
// Generic->AS casts go through integers: generic global == AS1 numerically;
// low 32 bits of a generic LDS address are the LDS byte offset (ISA flat rule).
__device__ __forceinline__ void copy16_g2l(const unsigned short* g, unsigned short* l) {
#if USE_ASYNC_LDS
    __builtin_amdgcn_global_load_async_to_lds_b128(
        (gvi4*)(uintptr_t)g,
        (lvi4*)(unsigned int)(uintptr_t)l,
        0, 0);
#else
    *(v8bf*)l = *(const v8bf*)g;
#endif
}

__device__ __forceinline__ void wait_stage_done() {
#if USE_ASYNC_LDS
    asm volatile("s_wait_asynccnt 0" ::: "memory");
#endif
}

// ---------------------------------------------------------------- pack X -> bf16
__global__ void pack_x_kernel(const float* __restrict__ x,
                              unsigned short* __restrict__ xb, int n) {
    int i = blockIdx.x * blockDim.x + threadIdx.x;
    if (i < n) xb[i] = f2bf(x[i]);
}

// ------------------------------------------------- pack W -> bf16, transposed (N,K)
__global__ void pack_wt_kernel(const float* __restrict__ w,
                               unsigned short* __restrict__ wt, int n) {
    int i = blockIdx.x * blockDim.x + threadIdx.x;   // over H_*NQKV_
    if (i >= n) return;
    int k = i / NQKV_;
    int c = i % NQKV_;
    wt[(size_t)c * H_ + k] = f2bf(w[i]);
}

// ------------------------------------------------- QKV GEMM (WMMA, LDS-staged)
// block = 256 threads = 8 waves (2 x 4), macro-tile 64 x 256, K-step 32.
// A/B staged to LDS via async DMA, double-buffered; each wave computes 32 x 64.
__global__ void __launch_bounds__(256)
qkv_gemm_kernel(const unsigned short* __restrict__ xb,
                const unsigned short* __restrict__ wt,
                float* __restrict__ qkv) {
    const int tid  = threadIdx.x;
    const int lane = tid & 31;
    const int wid  = tid >> 5;
    const int half = lane >> 4;
    const int r    = lane & 15;
    const int wr   = wid >> 2;       // 0..1  (M direction, 32 rows each)
    const int wc   = wid & 3;        // 0..3  (N direction, 64 cols each)

    const int m0 = (blockIdx.x / (NQKV_ / 256)) * 64;
    const int n0 = (blockIdx.x % (NQKV_ / 256)) * 256;

    __shared__ unsigned short As[2][64][32];    // 4 KB per buffer
    __shared__ unsigned short Bs[2][256][32];   // 16 KB per buffer

    // staging assignment: 16B chunks; A: 256 chunks (1/thread), B: 1024 (4/thread)
    const int srow = tid >> 2;       // 0..63
    const int sc8  = (tid & 3) * 8;  // element offset of 16B chunk

    v8f acc[2][4];
    #pragma unroll
    for (int mf = 0; mf < 2; ++mf)
        #pragma unroll
        for (int j = 0; j < 4; ++j)
            acc[mf][j] = (v8f){0, 0, 0, 0, 0, 0, 0, 0};

    // prologue: stage k0 = 0 into buffer 0
    {
        const int k0 = 0;
        copy16_g2l(xb + (size_t)(m0 + srow) * H_ + k0 + sc8, &As[0][srow][sc8]);
        #pragma unroll
        for (int j = 0; j < 4; ++j)
            copy16_g2l(wt + (size_t)(n0 + srow + j * 64) * H_ + k0 + sc8,
                       &Bs[0][srow + j * 64][sc8]);
    }

    const int nsteps = H_ / 32;   // 128
    for (int ks = 0; ks < nsteps; ++ks) {
        const int cur = ks & 1;
        wait_stage_done();        // this wave's DMA for buf[cur] has landed
        __syncthreads();          // everyone's has; everyone done reading buf[cur^1]

        if (ks + 1 < nsteps) {    // overlap next stage with compute
            const int k0 = (ks + 1) * 32;
            const int nxt = cur ^ 1;
            copy16_g2l(xb + (size_t)(m0 + srow) * H_ + k0 + sc8, &As[nxt][srow][sc8]);
            #pragma unroll
            for (int j = 0; j < 4; ++j)
                copy16_g2l(wt + (size_t)(n0 + srow + j * 64) * H_ + k0 + sc8,
                           &Bs[nxt][srow + j * 64][sc8]);
        }

        v16bf a0 = load_a16(&As[cur][wr * 32 + r][0], half);
        v16bf a1 = load_a16(&As[cur][wr * 32 + 16 + r][0], half);
        #pragma unroll
        for (int j = 0; j < 4; ++j) {
            v16bf b = *(const v16bf*)(&Bs[cur][wc * 64 + j * 16 + r][half * 16]);
            acc[0][j] = __builtin_amdgcn_wmma_f32_16x16x32_bf16(false, a0, false, b, (short)0, acc[0][j], false, false);
            acc[1][j] = __builtin_amdgcn_wmma_f32_16x16x32_bf16(false, a1, false, b, (short)0, acc[1][j], false, false);
        }
    }

    #pragma unroll
    for (int mf = 0; mf < 2; ++mf)
        #pragma unroll
        for (int i = 0; i < 8; ++i) {
            size_t row = (size_t)(m0 + wr * 32 + mf * 16 + i + half * 8) * NQKV_;
            #pragma unroll
            for (int j = 0; j < 4; ++j)
                qkv[row + n0 + wc * 64 + j * 16 + r] = acc[mf][j][i];
        }
}

// ----------------------------------------------- RoPE + repack to attention layout
// qb: (B, NH, S, HD) bf16 | kb: (B, NKV, S, HD) bf16 | vt: (B, NKV, HD, S) bf16
__global__ void rope_pack_kernel(const float* __restrict__ qkv,
                                 const float* __restrict__ cosb,
                                 const float* __restrict__ sinb,
                                 unsigned short* __restrict__ qb,
                                 unsigned short* __restrict__ kb,
                                 unsigned short* __restrict__ vt) {
    const size_t idx = (size_t)blockIdx.x * blockDim.x + threadIdx.x;
    if (idx >= (size_t)NTOK_ * NQKV_) return;
    const int col = (int)(idx % NQKV_);
    const int t   = (int)(idx / NQKV_);
    const int b   = t / S_;
    const int s   = t % S_;
    const int d   = col & (HD_ - 1);
    const float x = qkv[idx];

    if (col < (NH_ + NKV_) * HD_) {  // Q and K get RoPE
        const float cs = cosb[((size_t)b * S_ + s) * HD_ + d];
        const float sn = sinb[((size_t)b * S_ + s) * HD_ + d];
        const int   pr = (d < 64) ? (col + 64) : (col - 64);
        const float ot = qkv[(size_t)t * NQKV_ + pr];
        const float rh = (d < 64) ? -ot : ot;
        const unsigned short bv = f2bf(x * cs + rh * sn);
        if (col < NH_ * HD_) {
            const int hh = col >> 7;
            qb[(((size_t)b * NH_ + hh) * S_ + s) * HD_ + d] = bv;
        } else {
            const int kvh = (col - NH_ * HD_) >> 7;
            kb[(((size_t)b * NKV_ + kvh) * S_ + s) * HD_ + d] = bv;
        }
    } else {                          // V: transpose to (HD, S) for PV B-operand
        const int kvh = (col - (NH_ + NKV_) * HD_) >> 7;
        vt[(((size_t)b * NKV_ + kvh) * HD_ + d) * S_ + s] = f2bf(x);
    }
}

// ---------------------------------------------------------- flash attention (WMMA)
// one wave: 16 query rows x HD=128, online softmax over 32 keys per iteration
__global__ void flash_attn_kernel(const unsigned short* __restrict__ qb,
                                  const unsigned short* __restrict__ kb,
                                  const unsigned short* __restrict__ vt,
                                  float* __restrict__ out) {
    const int lane = threadIdx.x & 31;
    const int wid  = threadIdx.x >> 5;
    const int half = lane >> 4;
    const int r    = lane & 15;
    __shared__ unsigned short s_p[8][16][32];   // per-wave P staging (C->A transpose)

    const int wg  = blockIdx.x * 8 + wid;
    const int qt  = wg % (S_ / 16);
    const int h   = (wg / (S_ / 16)) % NH_;
    const int b   = wg / (NH_ * (S_ / 16));
    const int kvh = h / GROUPS_;
    const int q0  = qt * 16;

    const unsigned short* Qp = qb + ((size_t)(b * NH_ + h) * S_ + q0 + r) * HD_;
    const unsigned short* Kp = kb + (size_t)(b * NKV_ + kvh) * S_ * HD_;
    const unsigned short* Vp = vt + (size_t)(b * NKV_ + kvh) * HD_ * S_;

    v16bf aq[4];
    #pragma unroll
    for (int c = 0; c < 4; ++c) aq[c] = load_a16(Qp + c * 32, half);

    v8f acc[8];
    float mrow[8], lrow[8];
    #pragma unroll
    for (int i = 0; i < 8; ++i) {
        acc[i] = (v8f){0, 0, 0, 0, 0, 0, 0, 0};
        mrow[i] = -1e30f;
        lrow[i] = 0.f;
    }

    const float scale = 0.088388347648318447f;  // 1/sqrt(HD)
    const int ktot = q0 + 16;
    const int nch  = (ktot + 31) >> 5;

    for (int ch = 0; ch < nch; ++ch) {
        const int kbase = ch * 32;
        v8f sc[2];
        #pragma unroll
        for (int sub = 0; sub < 2; ++sub) {
            v8f s = {0, 0, 0, 0, 0, 0, 0, 0};
            const unsigned short* kp = Kp + (size_t)(kbase + sub * 16 + r) * HD_ + half * 16;
            #pragma unroll
            for (int c = 0; c < 4; ++c) {
                v16bf bk = *(const v16bf*)(kp + c * 32);
                s = __builtin_amdgcn_wmma_f32_16x16x32_bf16(false, aq[c], false, bk, (short)0, s, false, false);
            }
            sc[sub] = s;
        }
        // scale + causal mask (C layout: elem i -> row i+8*half, lane&15 -> col)
        #pragma unroll
        for (int sub = 0; sub < 2; ++sub) {
            const int kcol = kbase + sub * 16 + r;
            #pragma unroll
            for (int i = 0; i < 8; ++i) {
                const int qrow = q0 + i + half * 8;
                float v = sc[sub][i] * scale;
                sc[sub][i] = (kcol <= qrow) ? v : -1e30f;
            }
        }
        // online softmax: row reductions across the 16 lanes of this half-wave
        float alpha[8], nm[8];
        #pragma unroll
        for (int i = 0; i < 8; ++i) {
            float rm = fmaxf(sc[0][i], sc[1][i]);
            rm = fmaxf(rm, __shfl_xor(rm, 1));
            rm = fmaxf(rm, __shfl_xor(rm, 2));
            rm = fmaxf(rm, __shfl_xor(rm, 4));
            rm = fmaxf(rm, __shfl_xor(rm, 8));
            const float m2 = fmaxf(mrow[i], rm);
            alpha[i] = __expf(mrow[i] - m2);
            nm[i] = m2;
            mrow[i] = m2;
        }
        #pragma unroll
        for (int sub = 0; sub < 2; ++sub)
            #pragma unroll
            for (int i = 0; i < 8; ++i)
                sc[sub][i] = __expf(sc[sub][i] - nm[i]);
        #pragma unroll
        for (int i = 0; i < 8; ++i) {
            float rs = sc[0][i] + sc[1][i];
            rs += __shfl_xor(rs, 1);
            rs += __shfl_xor(rs, 2);
            rs += __shfl_xor(rs, 4);
            rs += __shfl_xor(rs, 8);
            lrow[i] = lrow[i] * alpha[i] + rs;
        }
        #pragma unroll
        for (int oc = 0; oc < 8; ++oc)
            #pragma unroll
            for (int i = 0; i < 8; ++i)
                acc[oc][i] *= alpha[i];

        // P: C layout -> bf16 in LDS -> reload as 16x32 A fragment
        #pragma unroll
        for (int sub = 0; sub < 2; ++sub)
            #pragma unroll
            for (int i = 0; i < 8; ++i)
                s_p[wid][i + half * 8][sub * 16 + r] = f2bf(sc[sub][i]);
#if defined(__gfx1250__)
        asm volatile("s_wait_dscnt 0" ::: "memory");
#endif
        v16bf ap = load_a16(&s_p[wid][r][0], half);

        // PV: B operand from V^T is a contiguous 32B load per lane
        #pragma unroll
        for (int oc = 0; oc < 8; ++oc) {
            v16bf bv = *(const v16bf*)(Vp + (size_t)(oc * 16 + r) * S_ + kbase + half * 16);
            acc[oc] = __builtin_amdgcn_wmma_f32_16x16x32_bf16(false, ap, false, bv, (short)0, acc[oc], false, false);
        }
    }

    // epilogue: normalize and write (B, S, NH*HD) f32
    #pragma unroll
    for (int i = 0; i < 8; ++i) {
        const int qrow  = q0 + i + half * 8;
        const float inv = 1.0f / lrow[i];
        float* op = out + (size_t)(b * S_ + qrow) * (NH_ * HD_) + h * HD_ + r;
        #pragma unroll
        for (int oc = 0; oc < 8; ++oc)
            op[oc * 16] = acc[oc][i] * inv;
    }
}

extern "C" void kernel_launch(void* const* d_in, const int* in_sizes, int n_in,
                              void* d_out, int out_size, void* d_ws, size_t ws_size,
                              hipStream_t stream) {
    const float* hidden = (const float*)d_in[0];
    const float* cosb   = (const float*)d_in[1];
    const float* sinb   = (const float*)d_in[2];
    const float* w_qkv  = (const float*)d_in[3];
    float* out = (float*)d_out;

    char* w = (char*)d_ws;
    size_t off = 0;
    unsigned short* xb = (unsigned short*)(w + off); off += (size_t)NTOK_ * H_ * 2;
    unsigned short* wt = (unsigned short*)(w + off); off += (size_t)NQKV_ * H_ * 2;
    float* qkvf        = (float*)(w + off);          off += (size_t)NTOK_ * NQKV_ * 4;
    unsigned short* qb = (unsigned short*)(w + off); off += (size_t)B_ * NH_ * S_ * HD_ * 2;
    unsigned short* kb = (unsigned short*)(w + off); off += (size_t)B_ * NKV_ * S_ * HD_ * 2;
    unsigned short* vt = (unsigned short*)(w + off); off += (size_t)B_ * NKV_ * S_ * HD_ * 2;
    (void)ws_size; (void)in_sizes; (void)n_in; (void)out_size;

    {
        int n = NTOK_ * H_;
        pack_x_kernel<<<(n + 255) / 256, 256, 0, stream>>>(hidden, xb, n);
    }
    {
        int n = H_ * NQKV_;
        pack_wt_kernel<<<(n + 255) / 256, 256, 0, stream>>>(w_qkv, wt, n);
    }
    {
        int blocks = (NTOK_ / 64) * (NQKV_ / 256);   // 64 * 24 = 1536
        qkv_gemm_kernel<<<blocks, 256, 0, stream>>>(xb, wt, qkvf);
    }
    {
        long long n = (long long)NTOK_ * NQKV_;
        rope_pack_kernel<<<(unsigned)((n + 255) / 256), 256, 0, stream>>>(qkvf, cosb, sinb, qb, kb, vt);
    }
    {
        int waves = B_ * NH_ * (S_ / 16);            // 8192 waves, 8 per block
        flash_attn_kernel<<<waves / 8, 256, 0, stream>>>(qb, kb, vt, out);
    }
}
